// TemporalDownsampler_7705171329521
// MI455X (gfx1250) — matrix operand order repaired and checked by reading, not compiled, not run on hardware
//
#include <hip/hip_runtime.h>

// ---------------- types ----------------
typedef __bf16 bf16_t;
typedef __bf16 bf16x8  __attribute__((ext_vector_type(8)));
typedef __bf16 bf16x16 __attribute__((ext_vector_type(16)));
typedef float  f32x8   __attribute__((ext_vector_type(8)));

__device__ __forceinline__ f32x8 wmma_bf16(bf16x16 a, bf16x16 b, f32x8 c) {
    return __builtin_amdgcn_wmma_f32_16x16x32_bf16(false, a, false, b, (short)0, c, false, false);
}

// ---------------- problem constants ----------------
#define BB      512
#define TT      81
#define DD      1536
#define NCHUNK  20
#define NS      (BB * NCHUNK)      // 10240 samples
#define K1      4608               // 3*D  (x0|x1|x2)
#define K2      1536
#define GROUPS_ 32
#define GCH     48                 // channels per group
#define EPSF    1e-5f
#define ASTR    80                 // LDS row stride (halves): 160B -> 16B aligned, bank-staggered

// ---------------- conversion kernels ----------------
__global__ void cvt_x_kernel(const float* __restrict__ x, bf16_t* __restrict__ Xbf) {
    size_t idx = (size_t)blockIdx.x * 256 + threadIdx.x;
    if (idx >= (size_t)NS * K1) return;
    int n = (int)(idx / K1);
    int k = (int)(idx % K1);
    int s = k / DD;                // position within chunk (0..2)
    int d = k % DD;
    int b  = n / NCHUNK;
    int ch = n % NCHUNK;
    int t  = 1 + ch * 4 + s;
    Xbf[idx] = (bf16_t)x[((size_t)b * TT + t) * DD + d];
}

__global__ void cvt_w1_kernel(const float* __restrict__ w1, bf16_t* __restrict__ WT1) {
    size_t idx = (size_t)blockIdx.x * 256 + threadIdx.x;
    if (idx >= (size_t)DD * K1) return;
    int o = (int)(idx / K1);
    int k = (int)(idx % K1);
    int s = k / DD;
    int i = k % DD;
    // WT1[o][s*D + i] = w1[o][i][s]
    WT1[idx] = (bf16_t)w1[((size_t)o * DD + i) * 3 + s];
}

__global__ void cvt_w2_kernel(const float* __restrict__ w2, bf16_t* __restrict__ WT2) {
    size_t idx = (size_t)blockIdx.x * 256 + threadIdx.x;
    if (idx >= (size_t)DD * K2) return;
    int o = (int)(idx / K2);
    int i = (int)(idx % K2);
    // only tap 2 of w2 is ever used
    WT2[idx] = (bf16_t)w2[((size_t)o * DD + i) * 3 + 2];
}

// ---------------- GEMM1: conv1 (y0 & y1) ----------------
// A: Xbf [NS][4608]   B: WT1 [1536][4608]   out: Y [NS][2][1536] (f32, +bias)
__global__ __launch_bounds__(256) void gemm1_kernel(const bf16_t* __restrict__ A,
                                                    const bf16_t* __restrict__ W,
                                                    const float*  __restrict__ bias,
                                                    float*        __restrict__ Y) {
    __shared__ __align__(16) bf16_t As[64 * ASTR];

    const int tid  = threadIdx.x;
    const int lane = tid & 31;
    const int wave = tid >> 5;
    const int sb   = blockIdx.x * 64;                       // sample base
    const int ocol = blockIdx.y * 128 + wave * 16 + (lane & 15);
    const int khalf = (lane < 16) ? 0 : 16;                 // A frag K offset
    const int koffB = (lane < 16) ? 0 : 16;                 // B frag K offset
    const int mrow  = lane & 15;
    const int maddM = (lane < 16) ? 0 : 8;                  // C/D: M = maddM + r

    f32x8 acc1[4], acc0[4];
#pragma unroll
    for (int i = 0; i < 4; ++i)
#pragma unroll
        for (int r = 0; r < 8; ++r) { acc1[i][r] = 0.f; acc0[i][r] = 0.f; }

    const size_t wrow = (size_t)ocol * K1;
    const int fr = tid >> 3;                // 0..31
    const int fc = (tid & 7) * 8;           // 0..56

    // ---- phase 1: k in [0, 1536): x0 columns feed BOTH y1 (tap 0) and y0 (tap 2)
    for (int k = 0; k < DD; k += 64) {
        __syncthreads();
        *(bf16x8*)(As + fr * ASTR + fc) =
            *(const bf16x8*)(A + (size_t)(sb + fr) * K1 + k + fc);
        *(bf16x8*)(As + (fr + 32) * ASTR + fc) =
            *(const bf16x8*)(A + (size_t)(sb + fr + 32) * K1 + k + fc);
        __syncthreads();

#pragma unroll
        for (int kk = 0; kk < 64; kk += 32) {
            const int kcol = k + kk + koffB;
            bf16x16 bf1 = *(const bf16x16*)(W + wrow + kcol);            // tap-0 weights (y1)
            bf16x16 bf0 = *(const bf16x16*)(W + wrow + 2 * DD + kcol);   // tap-2 weights (y0)
#pragma unroll
            for (int mt = 0; mt < 4; ++mt) {
                const bf16_t* rp = As + (mt * 16 + mrow) * ASTR + kk;
                bf16x8 lo = *(const bf16x8*)(rp + khalf);
                bf16x8 hi = *(const bf16x8*)(rp + 8 + khalf);
                bf16x16 af = __builtin_shufflevector(lo, hi,
                    0,1,2,3,4,5,6,7,8,9,10,11,12,13,14,15);
                acc1[mt] = wmma_bf16(af, bf1, acc1[mt]);
                acc0[mt] = wmma_bf16(af, bf0, acc0[mt]);
            }
        }
    }

    // ---- phase 2: k in [1536, 4608): x1/x2 columns feed y1 only
    for (int k = DD; k < K1; k += 64) {
        __syncthreads();
        *(bf16x8*)(As + fr * ASTR + fc) =
            *(const bf16x8*)(A + (size_t)(sb + fr) * K1 + k + fc);
        *(bf16x8*)(As + (fr + 32) * ASTR + fc) =
            *(const bf16x8*)(A + (size_t)(sb + fr + 32) * K1 + k + fc);
        __syncthreads();

#pragma unroll
        for (int kk = 0; kk < 64; kk += 32) {
            bf16x16 bf1 = *(const bf16x16*)(W + wrow + k + kk + koffB);
#pragma unroll
            for (int mt = 0; mt < 4; ++mt) {
                const bf16_t* rp = As + (mt * 16 + mrow) * ASTR + kk;
                bf16x8 lo = *(const bf16x8*)(rp + khalf);
                bf16x8 hi = *(const bf16x8*)(rp + 8 + khalf);
                bf16x16 af = __builtin_shufflevector(lo, hi,
                    0,1,2,3,4,5,6,7,8,9,10,11,12,13,14,15);
                acc1[mt] = wmma_bf16(af, bf1, acc1[mt]);
            }
        }
    }

    const float bv = bias[ocol];
#pragma unroll
    for (int mt = 0; mt < 4; ++mt)
#pragma unroll
        for (int r = 0; r < 8; ++r) {
            const size_t grow = (size_t)(sb + mt * 16 + maddM + r);
            Y[grow * (2 * DD) + ocol]      = acc0[mt][r] + bv;   // position 0
            Y[grow * (2 * DD) + DD + ocol] = acc1[mt][r] + bv;   // position 1
        }
}

// ---------------- GN1 + SiLU (outputs position 0 only, bf16) ----------------
__global__ __launch_bounds__(256) void gn1_silu_kernel(const float* __restrict__ Y,
                                                       const float* __restrict__ gw,
                                                       const float* __restrict__ gb,
                                                       bf16_t* __restrict__ S0) {
    __shared__ float ssum[GROUPS_], ssq[GROUPS_];
    const int tid = threadIdx.x;
    if (tid < GROUPS_) { ssum[tid] = 0.f; ssq[tid] = 0.f; }
    __syncthreads();
    const size_t base = (size_t)blockIdx.x * (2 * DD);
    for (int i = tid; i < 2 * DD; i += 256) {
        float v = Y[base + i];
        int c = i & (DD - 1);
        int g = c / GCH;
        atomicAdd(&ssum[g], v);
        atomicAdd(&ssq[g], v * v);
    }
    __syncthreads();
    const size_t ob = (size_t)blockIdx.x * DD;
    for (int c = tid; c < DD; c += 256) {
        int g = c / GCH;
        float mean = ssum[g] * (1.f / 96.f);
        float var  = ssq[g]  * (1.f / 96.f) - mean * mean;
        float inv  = rsqrtf(var + EPSF);
        float v = Y[base + c];                        // position 0 only
        float h = (v - mean) * inv * gw[c] + gb[c];
        float s = h / (1.f + __expf(-h));
        S0[ob + c] = (bf16_t)s;
    }
}

// ---------------- GEMM2: conv2 (z = W2[:,:,2] @ s0 + b2) ----------------
__global__ __launch_bounds__(256) void gemm2_kernel(const bf16_t* __restrict__ A,
                                                    const bf16_t* __restrict__ W,
                                                    const float*  __restrict__ bias,
                                                    float*        __restrict__ Z) {
    __shared__ __align__(16) bf16_t As[64 * ASTR];

    const int tid  = threadIdx.x;
    const int lane = tid & 31;
    const int wave = tid >> 5;
    const int sb   = blockIdx.x * 64;
    const int ocol = blockIdx.y * 128 + wave * 16 + (lane & 15);
    const int khalf = (lane < 16) ? 0 : 16;
    const int koffB = (lane < 16) ? 0 : 16;
    const int mrow  = lane & 15;
    const int maddM = (lane < 16) ? 0 : 8;

    f32x8 acc[4];
#pragma unroll
    for (int i = 0; i < 4; ++i)
#pragma unroll
        for (int r = 0; r < 8; ++r) acc[i][r] = 0.f;

    const size_t wrow = (size_t)ocol * K2;
    const int fr = tid >> 3;
    const int fc = (tid & 7) * 8;

    for (int k = 0; k < K2; k += 64) {
        __syncthreads();
        *(bf16x8*)(As + fr * ASTR + fc) =
            *(const bf16x8*)(A + (size_t)(sb + fr) * K2 + k + fc);
        *(bf16x8*)(As + (fr + 32) * ASTR + fc) =
            *(const bf16x8*)(A + (size_t)(sb + fr + 32) * K2 + k + fc);
        __syncthreads();

#pragma unroll
        for (int kk = 0; kk < 64; kk += 32) {
            bf16x16 bfr = *(const bf16x16*)(W + wrow + k + kk + koffB);
#pragma unroll
            for (int mt = 0; mt < 4; ++mt) {
                const bf16_t* rp = As + (mt * 16 + mrow) * ASTR + kk;
                bf16x8 lo = *(const bf16x8*)(rp + khalf);
                bf16x8 hi = *(const bf16x8*)(rp + 8 + khalf);
                bf16x16 af = __builtin_shufflevector(lo, hi,
                    0,1,2,3,4,5,6,7,8,9,10,11,12,13,14,15);
                acc[mt] = wmma_bf16(af, bfr, acc[mt]);
            }
        }
    }

    const float bv = bias[ocol];
#pragma unroll
    for (int mt = 0; mt < 4; ++mt)
#pragma unroll
        for (int r = 0; r < 8; ++r) {
            const size_t grow = (size_t)(sb + mt * 16 + maddM + r);
            Z[grow * DD + ocol] = acc[mt][r] + bv;
        }
}

// ---------------- GN2 + SiLU -> scatter to output rows 1..20 ----------------
__global__ __launch_bounds__(256) void gn2_silu_kernel(const float* __restrict__ Z,
                                                       const float* __restrict__ gw,
                                                       const float* __restrict__ gb,
                                                       float* __restrict__ out) {
    __shared__ float ssum[GROUPS_], ssq[GROUPS_];
    const int tid = threadIdx.x;
    if (tid < GROUPS_) { ssum[tid] = 0.f; ssq[tid] = 0.f; }
    __syncthreads();
    const int n = blockIdx.x;
    const size_t base = (size_t)n * DD;
    for (int c = tid; c < DD; c += 256) {
        float v = Z[base + c];
        int g = c / GCH;
        atomicAdd(&ssum[g], v);
        atomicAdd(&ssq[g], v * v);
    }
    __syncthreads();
    const int b = n / NCHUNK, ch = n % NCHUNK;
    const size_t ob = ((size_t)b * (NCHUNK + 1) + 1 + ch) * DD;
    for (int c = tid; c < DD; c += 256) {
        int g = c / GCH;
        float mean = ssum[g] * (1.f / 48.f);
        float var  = ssq[g]  * (1.f / 48.f) - mean * mean;
        float inv  = rsqrtf(var + EPSF);
        float v = Z[base + c];
        float h = (v - mean) * inv * gw[c] + gb[c];
        out[ob + c] = h / (1.f + __expf(-h));
    }
}

// ---------------- token-0 pass-through ----------------
__global__ void first_copy_kernel(const float* __restrict__ x, float* __restrict__ out) {
    size_t idx = (size_t)blockIdx.x * 256 + threadIdx.x;
    if (idx >= (size_t)BB * DD) return;
    int b = (int)(idx / DD);
    int c = (int)(idx % DD);
    out[(size_t)b * (NCHUNK + 1) * DD + c] = x[(size_t)b * TT * DD + c];
}

// ---------------- launcher ----------------
extern "C" void kernel_launch(void* const* d_in, const int* in_sizes, int n_in,
                              void* d_out, int out_size, void* d_ws, size_t ws_size,
                              hipStream_t stream) {
    const float* x    = (const float*)d_in[0];
    const float* w1   = (const float*)d_in[1];
    const float* b1   = (const float*)d_in[2];
    const float* gn1w = (const float*)d_in[3];
    const float* gn1b = (const float*)d_in[4];
    const float* w2   = (const float*)d_in[5];
    const float* b2   = (const float*)d_in[6];
    const float* gn2w = (const float*)d_in[7];
    const float* gn2b = (const float*)d_in[8];
    float* out = (float*)d_out;

    // workspace layout (regions overlaid across pipeline stages):
    //   [0)                 Xbf  bf16 NS*4608  (94,371,840 B)  -> reused as S0 after GEMM1
    //   [94,371,840)        WT1  bf16 1536*4608 (14,155,776 B)
    //   [108,527,616)       WT2  bf16 1536*1536 ( 4,718,592 B)
    //   [113,246,208)       Y    f32  NS*3072  (125,829,120 B) -> reused as Z after GN1
    char* ws = (char*)d_ws;
    bf16_t* Xbf = (bf16_t*)ws;
    bf16_t* WT1 = (bf16_t*)(ws + 94371840ull);
    bf16_t* WT2 = (bf16_t*)(ws + 108527616ull);
    float*  Y   = (float*)(ws + 113246208ull);
    bf16_t* S0  = Xbf;   // Xbf dead after GEMM1
    float*  Z   = Y;     // Y dead after GN1

    const size_t nx  = (size_t)NS * K1;
    const size_t nw1 = (size_t)DD * K1;
    const size_t nw2 = (size_t)DD * K2;

    cvt_x_kernel <<<dim3((unsigned)((nx  + 255) / 256)), 256, 0, stream>>>(x,  Xbf);
    cvt_w1_kernel<<<dim3((unsigned)((nw1 + 255) / 256)), 256, 0, stream>>>(w1, WT1);
    cvt_w2_kernel<<<dim3((unsigned)((nw2 + 255) / 256)), 256, 0, stream>>>(w2, WT2);

    gemm1_kernel<<<dim3(NS / 64, DD / 128), 256, 0, stream>>>(Xbf, WT1, b1, Y);
    gn1_silu_kernel<<<dim3(NS), 256, 0, stream>>>(Y, gn1w, gn1b, S0);
    gemm2_kernel<<<dim3(NS / 64, DD / 128), 256, 0, stream>>>(S0, WT2, b2, Z);
    gn2_silu_kernel<<<dim3(NS), 256, 0, stream>>>(Z, gn2w, gn2b, out);

    first_copy_kernel<<<dim3((BB * DD + 255) / 256), 256, 0, stream>>>(x, out);
}
